// LatteAR_10307921510601
// MI455X (gfx1250) — matrix operand order repaired and checked by reading, not compiled, not run on hardware
//
#include <hip/hip_runtime.h>
#include <math.h>

typedef __bf16 bf16;
typedef __attribute__((ext_vector_type(16))) __bf16 v16bf;
typedef __attribute__((ext_vector_type(8)))  float  v8f;
typedef __attribute__((ext_vector_type(4)))  unsigned int v4u;
typedef __attribute__((ext_vector_type(8)))  int v8i;
typedef __attribute__((ext_vector_type(4)))  int v4i;

#define NB 8      // batch
#define NT 16     // time
#define PP 256    // pixels per image (16x16)
#define DT 384    // token dim
#define DS 512    // state dim
#define NO 8      // options
#define RD 32     // r dim
#define NC1 1024
#define NSTEP 15

#if defined(__HIP_DEVICE_COMPILE__) && defined(__gfx1250__) && \
    __has_builtin(__builtin_amdgcn_tensor_load_to_lds) && \
    __has_builtin(__builtin_amdgcn_s_wait_tensorcnt)
#define USE_TDM 1
#else
#define USE_TDM 0
#endif

__device__ __forceinline__ float geluf(float x) {
    return 0.5f * x * (1.0f + erff(x * 0.70710678118654752f));
}

// ---------------------------------------------------------------- embedding
__global__ __launch_bounds__(256) void embed_kernel(
    const int* __restrict__ y1, const int* __restrict__ y2,
    const float* __restrict__ e1, const float* __restrict__ e2,
    bf16* __restrict__ emb)   // layout [t][b][p][c], c=384
{
    const size_t total = (size_t)NT * NB * PP * DT;
    for (size_t i = (size_t)blockIdx.x * 256 + threadIdx.x; i < total;
         i += (size_t)gridDim.x * 256) {
        int c = (int)(i % DT);
        size_t r = i / DT;
        int p = (int)(r % PP);
        size_t r2 = r / PP;
        int b = (int)(r2 % NB);
        int t = (int)(r2 / NB);
        int idx = (b * NT + t) * PP + p;
        float v = e1[(size_t)y1[idx] * DT + c] + e2[(size_t)y2[idx] * DT + c];
        int yy = p >> 4, xx = p & 15;
        float pos;
        if (c < 192) {
            int j = c >> 1;
            float ang = (float)yy * expf(-(float)j * (9.2103403719761836f / 96.0f));
            pos = (c & 1) ? cosf(ang) : sinf(ang);
        } else {
            int j = (c - 192) >> 1;
            float ang = (float)xx * expf(-(float)j * (9.2103403719761836f / 96.0f));
            pos = (c & 1) ? cosf(ang) : sinf(ang);
        }
        emb[i] = (bf16)(v + pos);
    }
}

// spatial means: one block per (t,b)
__global__ __launch_bounds__(256) void mean_kernel(
    const bf16* __restrict__ emb, float* __restrict__ means)
{
    int tb = blockIdx.x;                       // 0..127
    const bf16* base = emb + (size_t)tb * PP * DT;
    for (int c = threadIdx.x; c < DT; c += 256) {
        float s = 0.f;
        for (int p = 0; p < PP; ++p) s += (float)base[(size_t)p * DT + c];
        means[(size_t)tb * DT + c] = s * (1.0f / 256.0f);
    }
}

// ------------------------------------------------- weight relayout to bf16
// in:  w[o][i][kh][kw]   out: [(kh*KS+kw)][i][o]
__global__ __launch_bounds__(256) void wconv_kernel(
    const float* __restrict__ w, bf16* __restrict__ out, int O, int I, int KS)
{
    size_t total = (size_t)O * I * KS * KS;
    for (size_t idx = (size_t)blockIdx.x * 256 + threadIdx.x; idx < total;
         idx += (size_t)gridDim.x * 256) {
        int kw = (int)(idx % KS);
        size_t t1 = idx / KS;
        int kh = (int)(t1 % KS);
        size_t t2 = t1 / KS;
        int ii = (int)(t2 % I);
        int o  = (int)(t2 / I);
        out[(size_t)(kh * KS + kw) * I * O + (size_t)ii * O + o] = (bf16)w[idx];
    }
}

// ----------------------------------------------------------- concat builder
__global__ __launch_bounds__(256) void concat_kernel(
    const bf16* __restrict__ emb_t, const float* __restrict__ s_state,
    bf16* __restrict__ xcat)
{
    const size_t total = (size_t)NB * PP * (DT + DS);
    for (size_t i = (size_t)blockIdx.x * 256 + threadIdx.x; i < total;
         i += (size_t)gridDim.x * 256) {
        int c = (int)(i % (DT + DS));
        size_t pix = i / (DT + DS);
        int b = (int)(pix >> 8);
        bf16 v;
        if (c < DT) v = emb_t[pix * DT + c];
        else        v = (bf16)s_state[b * DS + (c - DT)];
        xcat[i] = v;
    }
}

// -------------------------------------------------------- WMMA implicit GEMM
// in : [B][16][16][Cin] bf16 (NHWC)    wgt : [kk][Cin][Cout] bf16
// block: 256 thr (8 waves); tile 64 pixels x 64 Cout; grid (32, Cout/64)
// B operand tiles (64 k x 64 n) are streamed by the Tensor Data Mover with
// double buffering; A (pixel gather with conv halo) uses vector loads.

#if USE_TDM
__device__ __forceinline__ void tdm_load_b_tile(const bf16* gsrc,
                                                unsigned lds_off, int Cout)
{
    unsigned long long ga = (unsigned long long)(size_t)gsrc;
    v4u g0;
    g0[0] = 1u;                                     // count=1, user descriptor
    g0[1] = lds_off;                                // lds_addr (bytes)
    g0[2] = (unsigned)(ga & 0xffffffffu);           // global_addr[31:0]
    g0[3] = (unsigned)((ga >> 32) & 0x1ffffffu)     // global_addr[56:32]
          | 0x80000000u;                            // type=2 (image)
    // data_size=2B (code 1), pad_enable=1, pad_interval=32 DWORDs (code 4),
    // pad_amount=4 DWORDs (code 3)  -> LDS row stride 72 bf16
    unsigned dim0 = (unsigned)Cout;                 // elements per tensor row
    v8i g1;
    g1[0] = (int)0x07110000u;
    g1[1] = (int)((dim0 & 0xffffu) << 16);          // tensor_dim0 low16
    g1[2] = (int)((dim0 >> 16) | (64u << 16));      // dim0 hi16 | tensor_dim1=64
    g1[3] = (int)(64u << 16);                       // tile_dim0 = 64
    g1[4] = (int)64u;                               // tile_dim1 = 64, tile_dim2=0
    g1[5] = (int)dim0;                              // tensor_dim0_stride low32
    g1[6] = 0;                                      // stride0 hi | stride1 low
    g1[7] = 0;
    v4i z4 = {0, 0, 0, 0};
    v8i z8 = {0, 0, 0, 0, 0, 0, 0, 0};
    // clang-23 / therock-10.0 lane: 6-arg form
    __builtin_amdgcn_tensor_load_to_lds(g0, g1, z4, z4, z8, 0);
}
#endif

__global__ __launch_bounds__(256) void conv_wmma_kernel(
    const bf16* __restrict__ in, const bf16* __restrict__ wgt,
    const float* __restrict__ bias, void* __restrict__ out,
    int Cin, int Cout, int ksize, int act_gelu, int out_f32)
{
    __shared__ __align__(16) bf16 As[64 * 72];      // 64 pixels x 64 k
    __shared__ __align__(16) bf16 Bs[2][64 * 72];   // 64 k x 64 n, dbl-buffered

    const int tid   = threadIdx.x;
    const int lane  = tid & 31;
    const int wave  = tid >> 5;
    const int img   = blockIdx.x >> 2;      // batch index
    const int ytile = blockIdx.x & 3;       // rows [ytile*4, ytile*4+4)
    const int nblk  = blockIdx.y;           // 64-wide Cout block
    const int mt    = wave & 3;             // M tile 0..3
    const int ntb   = (wave >> 2) * 2;      // N tile base 0 or 2

    v8f acc0 = {}; v8f acc1 = {};
    const int pad     = ksize >> 1;
    const int cchunks = Cin >> 6;                  // 64-wide K chunks per tap
    const int nchunks = ksize * ksize * cchunks;

#if USE_TDM
    if (wave == 0) {
        const bf16* g0src = wgt + (size_t)nblk * 64;
        tdm_load_b_tile(g0src, (unsigned)(size_t)&Bs[0][0], Cout);
    }
#endif

    for (int ci = 0; ci < nchunks; ++ci) {
        const int kk = ci / cchunks;
        const int c0 = (ci - kk * cchunks) << 6;
        const int dy = kk / ksize - pad;
        const int dx = kk % ksize - pad;

        {   // stage A: 64 pixels x 64 channels, 32B per thread
            int row = tid >> 2;
            int seg = tid & 3;
            int py = ytile * 4 + (row >> 4) + dy;
            int px = (row & 15) + dx;
            uint4* dst = (uint4*)&As[row * 72 + seg * 16];
            if (py >= 0 && py < 16 && px >= 0 && px < 16) {
                const bf16* src =
                    in + ((size_t)(img * 256 + py * 16 + px) * Cin + c0 + seg * 16);
                dst[0] = ((const uint4*)src)[0];
                dst[1] = ((const uint4*)src)[1];
            } else {
                dst[0] = make_uint4(0u, 0u, 0u, 0u);
                dst[1] = make_uint4(0u, 0u, 0u, 0u);
            }
        }

#if USE_TDM
        if (wave == 0) {
            if (ci + 1 < nchunks) {
                int kk2 = (ci + 1) / cchunks;
                int c02 = ((ci + 1) - kk2 * cchunks) << 6;
                const bf16* nsrc =
                    wgt + ((size_t)kk2 * Cin + c02) * Cout + nblk * 64;
                tdm_load_b_tile(nsrc, (unsigned)(size_t)&Bs[(ci + 1) & 1][0], Cout);
                __builtin_amdgcn_s_wait_tensorcnt((short)1);  // tile ci done
            } else {
                __builtin_amdgcn_s_wait_tensorcnt((short)0);
            }
        }
        const bf16* Bsb = Bs[ci & 1];
#else
        {   // fallback: stage B with vector loads
            int r  = tid >> 2;
            int sg = tid & 3;
            const bf16* src =
                wgt + ((size_t)kk * Cin + c0 + r) * Cout + nblk * 64 + sg * 16;
            uint4* bd = (uint4*)&Bs[0][r * 72 + sg * 16];
            bd[0] = ((const uint4*)src)[0];
            bd[1] = ((const uint4*)src)[1];
        }
        const bf16* Bsb = Bs[0];
#endif
        __syncthreads();

        // fragments + 4 WMMAs (two K=32 sub-chunks x two N tiles)
        const int arow = (mt * 16 + (lane & 15)) * 72;
        const int kb   = (lane & 16) ? 16 : 0;
        const int ko   = (lane & 16) ? 16 : 0;
        const int nc0  = ntb * 16 + (lane & 15);
#pragma unroll
        for (int kc = 0; kc < 2; ++kc) {
            const bf16* ap = &As[arow + kc * 32];
            v16bf a;
#pragma unroll
            for (int e = 0; e < 16; ++e)
                a[e] = ap[kb + e];
            v16bf b0, b1;
#pragma unroll
            for (int e = 0; e < 16; ++e) {
                int krow = kc * 32 + ko + e;
                b0[e] = Bsb[krow * 72 + nc0];
                b1[e] = Bsb[krow * 72 + nc0 + 16];
            }
            acc0 = __builtin_amdgcn_wmma_f32_16x16x32_bf16(
                false, a, false, b0, (short)0, acc0, false, false);
            acc1 = __builtin_amdgcn_wmma_f32_16x16x32_bf16(
                false, a, false, b1, (short)0, acc1, false, false);
        }
        __syncthreads();
    }

    // epilogue: C/D layout -> VGPR r: M = r + (lane>=16 ? 8 : 0), N = lane&15
    int mrb = (lane & 16) ? 8 : 0;
    int ncl = lane & 15;
#pragma unroll
    for (int r = 0; r < 8; ++r) {
        int m = mt * 16 + mrb + r;                 // pixel within 64-block
        size_t pix = (size_t)img * 256 + ytile * 64 + m;
        for (int half = 0; half < 2; ++half) {
            float v = half ? acc1[r] : acc0[r];
            int n = nblk * 64 + (ntb + half) * 16 + ncl;
            v += bias[n];
            if (act_gelu) v = geluf(v);
            if (out_f32) ((float*)out)[pix * Cout + n] = v;
            else         ((bf16*)out)[pix * Cout + n] = (bf16)v;
        }
    }
}

// ------------------------------------------------------- small recurrent MLP
__device__ void dense8(const float* __restrict__ X, int I,
                       const float* __restrict__ W, const float* __restrict__ Bb,
                       float* __restrict__ Y, int O, int act)
{
    for (int idx = threadIdx.x; idx < NB * O; idx += 256) {
        int b = idx / O, o = idx - b * O;
        float acc = Bb ? Bb[o] : 0.f;
        const float* w = W + (size_t)o * I;
        const float* x = X + (size_t)b * I;
        for (int i = 0; i < I; ++i) acc = fmaf(x[i], w[i], acc);
        if (act == 1)      acc = geluf(acc);
        else if (act == 2) acc = tanhf(acc);
        else if (act == 3) acc = 1.f / (1.f + expf(-acc));
        Y[idx] = acc;
    }
    __syncthreads();
}

__device__ void softmax8(float* row) {
    float m = -1e30f;
    for (int k = 0; k < NO; ++k) m = fmaxf(m, row[k]);
    float s = 0.f;
    for (int k = 0; k < NO; ++k) { float e = expf(row[k] - m); row[k] = e; s += e; }
    float inv = 1.f / s;
    for (int k = 0; k < NO; ++k) row[k] *= inv;
}

__global__ __launch_bounds__(256) void mlp_kernel(
    const float* __restrict__ means, float* __restrict__ s_state,
    float* __restrict__ q_state,
    const float* Uw,
    const float* aPb, const float* aPw,
    const float* g1b, const float* g1w,
    const float* g2b, const float* g2w,
    const float* g3b, const float* g3w,
    const float* h1b, const float* h1w,
    const float* h2b, const float* h2w,
    const float* opt_embed,
    const float* pi1b, const float* pi1w,
    const float* pi2b, const float* pi2w,
    const float* po1b, const float* po1w,
    const float* po2b, const float* po2w,
    const float* uPb, const float* uPw,
    float* __restrict__ scr, float* __restrict__ lossAcc, int tprev, int tcur)
{
    __shared__ float sS[NB * DS];
    __shared__ float sUp[NB * DT];
    __shared__ float sUc[NB * DT];
    __shared__ float sOx[NB * DS];
    __shared__ float sQpost[NB * NO], sQprev[NB * NO], sPi[NB * NO];
    __shared__ float sHz[NB], sG[NB];

    const int tid = threadIdx.x;
    for (int i = tid; i < NB * DS; i += 256) sS[i] = s_state[i];
    for (int i = tid; i < NB * DT; i += 256) {
        sUp[i] = means[(size_t)tprev * NB * DT + i];
        sUc[i] = means[(size_t)tcur  * NB * DT + i];
    }
    for (int i = tid; i < NB * NO; i += 256) sQprev[i] = q_state[i];
    __syncthreads();

    float* bufCat = scr;                         // 8*1408
    float* bufA = bufCat + NB * (DS + DT + DS);  // 8*512
    float* bufB = bufA + NB * DS;
    float* bufC = bufB + NB * DS;

    // q_post = softmax(post2(gelu(post1(uc))))
    dense8(sUc, DT, po1w, po1b, bufA, DT, 1);
    dense8(bufA, DT, po2w, po2b, sQpost, NO, 0);
    if (tid < NB) softmax8(&sQpost[tid * NO]);
    __syncthreads();

    // hz = sigmoid(haz2(gelu(haz1([s,up]))))
    for (int i = tid; i < NB * (DS + DT); i += 256) {
        int b = i / (DS + DT), c = i % (DS + DT);
        bufCat[i] = (c < DS) ? sS[b * DS + c] : sUp[b * DT + (c - DS)];
    }
    __syncthreads();
    dense8(bufCat, DS + DT, h1w, h1b, bufA, DS / 2, 1);
    dense8(bufA, DS / 2, h2w, h2b, sHz, 1, 3);

    // pi = softmax(pi2(gelu(pi1(s))))
    dense8(sS, DS, pi1w, pi1b, bufA, DS, 1);
    dense8(bufA, DS, pi2w, pi2b, sPi, NO, 0);
    if (tid < NB) softmax8(&sPi[tid * NO]);
    __syncthreads();

    // prior, kl_opt, hz accumulation
    if (tid < NB) {
        int b = tid; float kl = 0.f; float hz = sHz[b];
        for (int k = 0; k < NO; ++k) {
            float prior = (1.f - hz) * sQprev[b * NO + k] + hz * sPi[b * NO + k];
            float qp = sQpost[b * NO + k];
            kl += qp * (logf(qp + 1e-8f) - logf(prior + 1e-8f));
        }
        atomicAdd(&lossAcc[2], kl * (1.f / (8.f * 15.f)));
        atomicAdd(&lossAcc[4], hz * (1.f / (8.f * 15.f)));
    }
    __syncthreads();

    // o_exp = q_post @ opt_embed
    for (int i = tid; i < NB * DS; i += 256) {
        int b = i >> 9, o = i & (DS - 1);
        float acc = 0.f;
        for (int k = 0; k < NO; ++k) acc += sQpost[b * NO + k] * opt_embed[k * DS + o];
        sOx[i] = acc;
    }
    __syncthreads();

    dense8(sOx, DS, aPw, aPb, bufA, DS, 2);   // a
    dense8(sOx, DS, uPw, uPb, bufB, DS, 2);   // u_scale
    dense8(sUp, DT, Uw, nullptr, bufC, DS, 0);// u_projv
    for (int i = tid; i < NB * DS; i += 256)  // s_bar (into bufA)
        bufA[i] = bufA[i] * sS[i] + bufB[i] * bufC[i];
    __syncthreads();

    // gate
    for (int i = tid; i < NB * (DS + DT + DS); i += 256) {
        int b = i / (DS + DT + DS), c = i % (DS + DT + DS);
        float v;
        if (c < DS)           v = sS[b * DS + c];
        else if (c < DS + DT) v = sUp[b * DT + (c - DS)];
        else                  v = sOx[b * DS + (c - DS - DT)];
        bufCat[i] = v;
    }
    __syncthreads();
    dense8(bufCat, DS + DT + DS, g1w, g1b, bufB, DS, 1);
    dense8(bufB, DS, g2w, g2b, bufC, DS, 1);
    dense8(bufC, DS, g3w, g3b, sG, 1, 3);

    for (int i = tid; i < NB * DS; i += 256) {
        int b = i >> 9; float g = sG[b];
        s_state[i] = g * bufA[i] + (1.f - g) * sS[i];
    }
    for (int i = tid; i < NB * NO; i += 256) q_state[i] = sQpost[i];
}

// ---------------------------------------------------------- CE (log_softmax)
__global__ __launch_bounds__(256) void ce_kernel(
    const float* __restrict__ logits, const int* __restrict__ yidx,
    int t, int C, float* __restrict__ lossAcc, int slot)
{
    __shared__ float red[256];
    const int tid = threadIdx.x;
    size_t pix = blockIdx.x;
    int b = (int)(pix >> 8), p = (int)(pix & 255);
    const float* row = logits + pix * C;
    int target = yidx[(b * NT + t) * PP + p];

    float m = -1e30f;
    for (int c = tid; c < C; c += 256) m = fmaxf(m, row[c]);
    red[tid] = m; __syncthreads();
    for (int s = 128; s > 0; s >>= 1) {
        if (tid < s) red[tid] = fmaxf(red[tid], red[tid + s]);
        __syncthreads();
    }
    m = red[0]; __syncthreads();

    float sum = 0.f;
    for (int c = tid; c < C; c += 256) sum += expf(row[c] - m);
    red[tid] = sum; __syncthreads();
    for (int s = 128; s > 0; s >>= 1) {
        if (tid < s) red[tid] += red[tid + s];
        __syncthreads();
    }
    if (tid == 0) {
        float lse = m + logf(red[0]);
        atomicAdd(&lossAcc[slot], (lse - row[target]) * (1.f / (2048.f * 15.f)));
    }
}

// -------------------------------------------------- rp_mu / rp_lv 1x1 heads
__global__ __launch_bounds__(256) void rp_head_kernel(
    const bf16* __restrict__ h,
    const float* __restrict__ muW, const float* __restrict__ muB,
    const float* __restrict__ lvW, const float* __restrict__ lvB,
    float* __restrict__ mup, float* __restrict__ lvp)
{
    int i = blockIdx.x * 256 + threadIdx.x;     // over 2048*32
    if (i >= NB * PP * RD) return;
    int c = i & 31;
    size_t pix = (size_t)(i >> 5);
    const bf16* x = h + pix * DT;
    const float* wm = muW + (size_t)c * DT;
    const float* wl = lvW + (size_t)c * DT;
    float am = muB[c], al = lvB[c];
    for (int k = 0; k < DT; ++k) {
        float xv = (float)x[k];
        am = fmaf(xv, wm[k], am);
        al = fmaf(xv, wl[k], al);
    }
    mup[i] = am; lvp[i] = al;
}

// ------------------------------------------------------------------ KL(r)
__global__ __launch_bounds__(256) void klr_kernel(
    const float* __restrict__ mup, const float* __restrict__ lvp,
    const float* __restrict__ rmu, const float* __restrict__ rlv,
    int t, float* __restrict__ lossAcc)
{
    __shared__ float red[256];
    const int tid = threadIdx.x;
    int i = blockIdx.x * 256 + tid;            // total 65536
    float term = 0.f;
    if (i < NB * PP * RD) {
        int c = i & 31;
        int pix = i >> 5;
        int b = pix >> 8, p = pix & 255;
        float mp = mup[i], lp = lvp[i];
        size_t qi = ((size_t)(b * NT + t) * RD + c) * PP + p;
        float mq = rmu[qi], lq = rlv[qi];
        float d = mq - mp;
        term = 0.5f * (expf(lp - lq) + d * d / (expf(lp) + 1e-8f) + lp - lq - 1.f);
    }
    red[tid] = term; __syncthreads();
    for (int s = 128; s > 0; s >>= 1) {
        if (tid < s) red[tid] += red[tid + s];
        __syncthreads();
    }
    if (tid == 0) atomicAdd(&lossAcc[3], red[0] * (1.f / (65536.f * 15.f)));
}

// ------------------------------------------------------------- init / final
__global__ void init_kernel(float* s_state, float* q_state, float* lossAcc) {
    int i = blockIdx.x * 256 + threadIdx.x;
    if (i < NB * DS) s_state[i] = 0.f;
    if (i < NB * NO) q_state[i] = ((i & 7) == 0) ? 1.f : 0.f;
    if (i < 16)      lossAcc[i] = 0.f;
}

__global__ void final_kernel(const float* lossAcc, float* out) {
    if (threadIdx.x == 0 && blockIdx.x == 0)
        out[0] = lossAcc[0] + lossAcc[1] + lossAcc[2]
               + 0.1f * lossAcc[3] + 0.01f * lossAcc[4];
}

// ================================================================== launch
extern "C" void kernel_launch(void* const* d_in, const int* in_sizes, int n_in,
                              void* d_out, int out_size, void* d_ws, size_t ws_size,
                              hipStream_t stream)
{
    (void)in_sizes; (void)n_in; (void)out_size; (void)ws_size;
    const int*   y1  = (const int*)d_in[0];
    const int*   y2  = (const int*)d_in[1];
    const float* rmu = (const float*)d_in[2];
    const float* rlv = (const float*)d_in[3];
    // params flattened alphabetically, 'b' before 'w' within each dict
    int k = 4;
    const float* Uw    = (const float*)d_in[k++]; // U.w           (512,384)
    const float* aPb   = (const float*)d_in[k++]; // a_proj.b
    const float* aPw   = (const float*)d_in[k++]; // a_proj.w      (512,512)
    const float* e1    = (const float*)d_in[k++]; // e1            (1024,384)
    const float* e2    = (const float*)d_in[k++]; // e2            (1024,384)
    const float* g1b   = (const float*)d_in[k++];
    const float* g1w   = (const float*)d_in[k++]; // gate1.w       (512,1408)
    const float* g2b   = (const float*)d_in[k++];
    const float* g2w   = (const float*)d_in[k++]; // gate2.w       (512,512)
    const float* g3b   = (const float*)d_in[k++];
    const float* g3w   = (const float*)d_in[k++]; // gate3.w       (1,512)
    const float* h1b   = (const float*)d_in[k++];
    const float* h1w   = (const float*)d_in[k++]; // haz1.w        (256,896)
    const float* h2b   = (const float*)d_in[k++];
    const float* h2w   = (const float*)d_in[k++]; // haz2.w        (1,256)
    const float* optE  = (const float*)d_in[k++]; // opt_embed     (8,512)
    const float* pi1b  = (const float*)d_in[k++];
    const float* pi1w  = (const float*)d_in[k++]; // pi1.w         (512,512)
    const float* pi2b  = (const float*)d_in[k++];
    const float* pi2w  = (const float*)d_in[k++]; // pi2.w         (8,512)
    const float* po1b  = (const float*)d_in[k++];
    const float* po1w  = (const float*)d_in[k++]; // post1.w       (384,384)
    const float* po2b  = (const float*)d_in[k++];
    const float* po2w  = (const float*)d_in[k++]; // post2.w       (8,384)
    const float* roc1b = (const float*)d_in[k++];
    const float* roc1w = (const float*)d_in[k++]; // ro_c1.w       (384,896,3,3)
    const float* roc2b = (const float*)d_in[k++];
    const float* roc2w = (const float*)d_in[k++]; // ro_c2.w       (384,384,3,3)
    const float* roh1b = (const float*)d_in[k++];
    const float* roh1w = (const float*)d_in[k++]; // ro_h1.w       (1024,384,1,1)
    const float* roh2b = (const float*)d_in[k++];
    const float* roh2w = (const float*)d_in[k++]; // ro_h2.w       (1024,384,1,1)
    const float* rpc1b = (const float*)d_in[k++];
    const float* rpc1w = (const float*)d_in[k++]; // rp_c1.w       (384,896,3,3)
    const float* rpc2b = (const float*)d_in[k++];
    const float* rpc2w = (const float*)d_in[k++]; // rp_c2.w       (384,384,3,3)
    const float* rplvb = (const float*)d_in[k++];
    const float* rplvw = (const float*)d_in[k++]; // rp_lv.w       (32,384,1,1)
    const float* rpmub = (const float*)d_in[k++];
    const float* rpmuw = (const float*)d_in[k++]; // rp_mu.w       (32,384,1,1)
    const float* uPb   = (const float*)d_in[k++];
    const float* uPw   = (const float*)d_in[k++]; // u_proj.w      (512,512)

    // ---- workspace layout
    char* ws = (char*)d_ws;
    size_t off = 0;
    auto alloc = [&](size_t bytes) -> char* {
        char* p = ws + off;
        off += (bytes + 255) & ~(size_t)255;
        return p;
    };
    bf16*  emb    = (bf16*) alloc((size_t)NT * NB * PP * DT * 2);   // 25.2 MB
    float* means  = (float*)alloc((size_t)NT * NB * DT * 4);
    float* sSt    = (float*)alloc(NB * DS * 4);
    float* qSt    = (float*)alloc(NB * NO * 4);
    bf16*  xcat1  = (bf16*) alloc((size_t)NB * PP * (DT + DS) * 2);
    bf16*  xcat2  = (bf16*) alloc((size_t)NB * PP * (DT + DS) * 2);
    bf16*  hA     = (bf16*) alloc((size_t)NB * PP * DT * 2);
    bf16*  hB     = (bf16*) alloc((size_t)NB * PP * DT * 2);
    float* logits = (float*)alloc((size_t)NB * PP * NC1 * 4);
    float* mup    = (float*)alloc((size_t)NB * PP * RD * 4);
    float* lvp    = (float*)alloc((size_t)NB * PP * RD * 4);
    bf16*  wRoc1  = (bf16*) alloc((size_t)9 * 896 * 384 * 2);
    bf16*  wRoc2  = (bf16*) alloc((size_t)9 * 384 * 384 * 2);
    bf16*  wRoh1  = (bf16*) alloc((size_t)384 * 1024 * 2);
    bf16*  wRoh2  = (bf16*) alloc((size_t)384 * 1024 * 2);
    bf16*  wRpc1  = (bf16*) alloc((size_t)9 * 896 * 384 * 2);
    bf16*  wRpc2  = (bf16*) alloc((size_t)9 * 384 * 384 * 2);
    float* scr    = (float*)alloc((size_t)NB * 2944 * 4);
    float* lossA  = (float*)alloc(64);

    // ---- prologue
    init_kernel<<<16, 256, 0, stream>>>(sSt, qSt, lossA);
    embed_kernel<<<49152, 256, 0, stream>>>(y1, y2, e1, e2, emb);
    mean_kernel<<<NT * NB, 256, 0, stream>>>(emb, means);
    wconv_kernel<<<12096, 256, 0, stream>>>(roc1w, wRoc1, 384, 896, 3);
    wconv_kernel<<<5184,  256, 0, stream>>>(roc2w, wRoc2, 384, 384, 3);
    wconv_kernel<<<1536,  256, 0, stream>>>(roh1w, wRoh1, 1024, 384, 1);
    wconv_kernel<<<1536,  256, 0, stream>>>(roh2w, wRoh2, 1024, 384, 1);
    wconv_kernel<<<12096, 256, 0, stream>>>(rpc1w, wRpc1, 384, 896, 3);
    wconv_kernel<<<5184,  256, 0, stream>>>(rpc2w, wRpc2, 384, 384, 3);

    // ---- sequential scan, 15 steps
    for (int step = 0; step < NSTEP; ++step) {
        const int tprev = step, tcur = step + 1;

        mlp_kernel<<<1, 256, 0, stream>>>(
            means, sSt, qSt, Uw, aPb, aPw, g1b, g1w, g2b, g2w, g3b, g3w,
            h1b, h1w, h2b, h2w, optE, pi1b, pi1w, pi2b, pi2w,
            po1b, po1w, po2b, po2w, uPb, uPw, scr, lossA, tprev, tcur);

        const bf16* embP = emb + (size_t)tprev * NB * PP * DT;
        const bf16* embC = emb + (size_t)tcur  * NB * PP * DT;
        concat_kernel<<<7168, 256, 0, stream>>>(embP, sSt, xcat1);
        concat_kernel<<<7168, 256, 0, stream>>>(embC, sSt, xcat2);

        // observation branch
        conv_wmma_kernel<<<dim3(32, 6), 256, 0, stream>>>(
            xcat1, wRoc1, roc1b, hA, 896, 384, 3, 1, 0);
        conv_wmma_kernel<<<dim3(32, 6), 256, 0, stream>>>(
            hA, wRoc2, roc2b, hB, 384, 384, 3, 1, 0);
        conv_wmma_kernel<<<dim3(32, 16), 256, 0, stream>>>(
            hB, wRoh1, roh1b, logits, 384, 1024, 1, 0, 1);
        ce_kernel<<<NB * PP, 256, 0, stream>>>(logits, y1, tcur, NC1, lossA, 0);
        conv_wmma_kernel<<<dim3(32, 16), 256, 0, stream>>>(
            hB, wRoh2, roh2b, logits, 384, 1024, 1, 0, 1);
        ce_kernel<<<NB * PP, 256, 0, stream>>>(logits, y2, tcur, NC1, lossA, 1);

        // prior branch
        conv_wmma_kernel<<<dim3(32, 6), 256, 0, stream>>>(
            xcat2, wRpc1, rpc1b, hA, 896, 384, 3, 1, 0);
        conv_wmma_kernel<<<dim3(32, 6), 256, 0, stream>>>(
            hA, wRpc2, rpc2b, hB, 384, 384, 3, 1, 0);
        rp_head_kernel<<<256, 256, 0, stream>>>(hB, rpmuw, rpmub, rplvw, rplvb, mup, lvp);
        klr_kernel<<<256, 256, 0, stream>>>(mup, lvp, rmu, rlv, tcur, lossA);
    }

    final_kernel<<<1, 32, 0, stream>>>(lossA, (float*)d_out);
}
